// MultiModalBIAttenNetworkLayers_43997644980650
// MI455X (gfx1250) — compile-verified, matrix operand order checked by reading, not run on hardware
//
#include <hip/hip_runtime.h>
#include <hip/hip_bf16.h>
#include <math.h>

// ---------------- CDNA5 / gfx1250 GEMM-heavy implementation -----------------
// wave32; WMMA 16x16x32 bf16 (fp32 accum) for all GEMMs; B-tile staged with
// GLOBAL_LOAD_ASYNC_TO_LDS_B32 (ASYNCcnt) -- CDNA5 async copy path.

#define HDIM 8
#define BM 128
#define BN 64
#define KT 32
#define LPAD 34            // bf16 LDS row stride (u16) -> conflict-free frag reads
#define BPAD 33            // f32 LDS row stride (dwords) for async-staged B tile
#define HSZ (1u << 19)
#define HMASK (HSZ - 1u)
#define INV_TEMP 0.17677669529663687f   // 1/sqrt(32)

typedef __attribute__((ext_vector_type(16))) __bf16 bf16x16;
typedef __attribute__((ext_vector_type(2)))  __bf16 bf16x2;
typedef __attribute__((ext_vector_type(8)))  float  f32x8;

struct GemmDesc {
  // Up to 4 A-segments, each `segw` columns wide; row width of each source == segw.
  const float* A0; const float* A1; const float* A2; const float* A3;
  const int*   I0; const int*   I1; const int*   I2; const int*   I3; // gather idx (<0 => zero row)
  const float* B;     // [K][N] row-major, or [N][K] if btrans (chan-MLP weights W[o][c])
  const float* bias;  // [N] or nullptr
  float*       C;
  int M, N, K, segw;
  int hstride;        // 1: rows are (e,h) pairs, elem = A[(e*segw+c)*8+h]
  int outh;           // 1: output [E,N,8] head-strided
  int btrans;         // B given as [N][K]
  int act;            // 0 none, 1 relu, 2 sigmoid
};

__device__ __forceinline__ unsigned short f2bf(float f) {
  unsigned u = __float_as_uint(f);
  unsigned r = 0x7fffu + ((u >> 16) & 1u);   // round-to-nearest-even
  return (unsigned short)((u + r) >> 16);
}

__device__ __forceinline__ float fetchA(const GemmDesc& d, int row, int k) {
  int s = k / d.segw, c = k - s * d.segw;
  const float* a; const int* ix;
  switch (s) {
    case 0:  a = d.A0; ix = d.I0; break;
    case 1:  a = d.A1; ix = d.I1; break;
    case 2:  a = d.A2; ix = d.I2; break;
    default: a = d.A3; ix = d.I3; break;
  }
  if (d.hstride) {
    int e = row >> 3, h = row & 7;
    if (ix) { int g = ix[e]; if (g < 0) return 0.f; e = g; }
    return a[((long long)e * d.segw + c) * HDIM + h];
  } else {
    int r = row;
    if (ix) { int g = ix[row]; if (g < 0) return 0.f; r = g; }
    return a[(long long)r * d.segw + c];
  }
}

// A-frag (16x32 bf16) per ISA 7.12.2: lanes 0-15 row m, lanes 16-31 same m,
// VGPR v<4: K = 2v + 8g (+pair), v>=4: K = 16 + 2(v-4) + 8g. B mirrored (N in lanes).
__device__ __forceinline__ bf16x16 loadFragA(const unsigned short* S, int tileRow, int lane) {
  const unsigned short* base = S + (tileRow + (lane & 15)) * LPAD;
  int g8 = (lane >> 4) * 8;
  bf16x16 f;
#pragma unroll
  for (int v = 0; v < 8; ++v) {
    int k0 = (v < 4) ? (2 * v + g8) : (16 + 2 * (v - 4) + g8);
    unsigned u = *(const unsigned*)(base + k0);       // dword-aligned (k0 even, LPAD even)
    bf16x2 p = __builtin_bit_cast(bf16x2, u);
    f[2 * v] = p[0]; f[2 * v + 1] = p[1];
  }
  return f;
}

// B-frag from f32 LDS tile (async-staged): convert pairs to packed bf16.
__device__ __forceinline__ bf16x16 loadFragB(const float* S, int tileRow, int lane) {
  const float* base = S + (tileRow + (lane & 15)) * BPAD;
  int g8 = (lane >> 4) * 8;
  bf16x16 f;
#pragma unroll
  for (int v = 0; v < 8; ++v) {
    int k0 = (v < 4) ? (2 * v + g8) : (16 + 2 * (v - 4) + g8);
    unsigned u = (unsigned)f2bf(base[k0]) | ((unsigned)f2bf(base[k0 + 1]) << 16);
    bf16x2 p = __builtin_bit_cast(bf16x2, u);
    f[2 * v] = p[0]; f[2 * v + 1] = p[1];
  }
  return f;
}

__global__ __launch_bounds__(256) void gemm_bf16_wmma(GemmDesc d) {
  __shared__ unsigned short As[BM * LPAD];
  __shared__ float          Bsf[BN * BPAD];
  const int tid = threadIdx.x, lane = tid & 31, wid = tid >> 5;
  const int mq = wid >> 1, nq = wid & 1;          // 4x2 wave grid
  const int rowBase = blockIdx.x * BM;
  const int colBase = blockIdx.y * BN;
  f32x8 acc[2][2] = {};

  for (int kk = 0; kk < d.K; kk += KT) {
    __syncthreads();
    // ---- stage A: 128x32 f32 -> bf16 (coalesced: lane -> k, row groups of 8)
    for (int j = 0; j < (BM * KT) / 256; ++j) {
      int i = tid + 256 * j;
      int r = i >> 5, k = i & 31;
      int grow = rowBase + r;
      float v = (grow < d.M) ? fetchA(d, grow, kk + k) : 0.f;
      As[r * LPAD + k] = f2bf(v);
    }
    // ---- stage B tile via CDNA5 async copy to LDS (ASYNCcnt-tracked).
    // n-major f32 tile; OOB columns clamped (their outputs are never stored).
#pragma unroll
    for (int j = 0; j < (BN * KT) / 256; ++j) {
      int i = tid + 256 * j;
      int n = i >> 5, k = i & 31;
      int kg = kk + k;                               // K is a multiple of 32
      int ng = colBase + n; if (ng >= d.N) ng = d.N - 1;
      const float* gp = d.btrans ? &d.B[(long long)ng * d.K + kg]
                                 : &d.B[(long long)kg * d.N + ng];
      unsigned loff = (unsigned)(size_t)(&Bsf[n * BPAD + k]);  // low 32b of generic ptr = LDS byte addr
      unsigned long long ga = (unsigned long long)(size_t)gp;
      asm volatile("global_load_async_to_lds_b32 %0, %1, off"
                   :: "v"(loff), "v"(ga) : "memory");
    }
    // prefetch next weight K-slab into L2 (global_prefetch_b8)
    if (!d.btrans && tid < 32 && kk + KT < d.K)
      __builtin_prefetch(&d.B[(long long)(kk + KT + tid) * d.N + colBase], 0, 3);
    asm volatile("s_wait_asynccnt 0x0" ::: "memory");
    __syncthreads();

    bf16x16 af0 = loadFragA(As,  mq * 32,      lane);
    bf16x16 af1 = loadFragA(As,  mq * 32 + 16, lane);
    bf16x16 bf0 = loadFragB(Bsf, nq * 32,      lane);
    bf16x16 bf1 = loadFragB(Bsf, nq * 32 + 16, lane);
    acc[0][0] = __builtin_amdgcn_wmma_f32_16x16x32_bf16(false, af0, false, bf0, (short)0, acc[0][0], false, false);
    acc[0][1] = __builtin_amdgcn_wmma_f32_16x16x32_bf16(false, af0, false, bf1, (short)0, acc[0][1], false, false);
    acc[1][0] = __builtin_amdgcn_wmma_f32_16x16x32_bf16(false, af1, false, bf0, (short)0, acc[1][0], false, false);
    acc[1][1] = __builtin_amdgcn_wmma_f32_16x16x32_bf16(false, af1, false, bf1, (short)0, acc[1][1], false, false);
  }

  // C/D layout: n = lane%16, m = vgpr + 8*(lane/16)
#pragma unroll
  for (int mt = 0; mt < 2; ++mt)
#pragma unroll
    for (int nt = 0; nt < 2; ++nt) {
      int ng = colBase + nq * 32 + nt * 16 + (lane & 15);
#pragma unroll
      for (int r = 0; r < 8; ++r) {
        int mrow = rowBase + mq * 32 + mt * 16 + r + 8 * (lane >> 4);
        if (mrow < d.M && ng < d.N) {
          float v = acc[mt][nt][r];
          if (d.bias) v += d.bias[ng];
          if (d.act == 1) v = fmaxf(v, 0.f);
          else if (d.act == 2) v = 1.f / (1.f + expf(-v));
          if (d.outh) {
            int e = mrow >> 3, h = mrow & 7;
            d.C[((long long)e * d.N + ng) * HDIM + h] = v;
          } else {
            d.C[(long long)mrow * d.N + ng] = v;
          }
        }
      }
    }
}

// ------------------------- auxiliary kernels --------------------------------

__global__ void fill_u32_k(unsigned* p, long long n, unsigned v) {
  long long i = blockIdx.x * 256LL + threadIdx.x; if (i < n) p[i] = v;
}
__global__ void fill_f32_k(float* p, long long n, float v) {
  long long i = blockIdx.x * 256LL + threadIdx.x; if (i < n) p[i] = v;
}

// deterministic reverse-edge map: key -> min edge id (matches stable argsort + searchsorted-left)
__global__ void hash_insert_k(const int* row, const int* col, unsigned* tk, unsigned* tv, int Nn, int E) {
  int e = blockIdx.x * 256 + threadIdx.x; if (e >= E) return;
  unsigned key = (unsigned)row[e] * (unsigned)Nn + (unsigned)col[e];
  unsigned h = (key * 2654435761u) & HMASK;
  for (;;) {
    unsigned prev = atomicCAS(&tk[h], 0xFFFFFFFFu, key);
    if (prev == 0xFFFFFFFFu || prev == key) { atomicMin(&tv[h], (unsigned)e); return; }
    h = (h + 1) & HMASK;
  }
}
__global__ void hash_lookup_k(const int* row, const int* col, const unsigned* tk, const unsigned* tv,
                              int* rev, int Nn, int E) {
  int e = blockIdx.x * 256 + threadIdx.x; if (e >= E) return;
  unsigned key = (unsigned)col[e] * (unsigned)Nn + (unsigned)row[e];
  unsigned h = (key * 2654435761u) & HMASK;
  int res = -1;
  for (unsigned p = 0; p < HSZ; ++p) {
    unsigned k = tk[h];
    if (k == 0xFFFFFFFFu) break;
    if (k == key) { res = (int)tv[h]; break; }
    h = (h + 1) & HMASK;
  }
  rev[e] = res;
}

__global__ void dmask_k(const float* pos, const int* row, const int* col,
                        const float* Wd1, const float* bd1, const float* Wd2, const float* bd2,
                        float* dmask, int E) {
  int e = blockIdx.x * 256 + threadIdx.x; if (e >= E) return;
  int i = row[e], j = col[e];
  float dx = pos[i * 3 + 0] - pos[j * 3 + 0];
  float dy = pos[i * 3 + 1] - pos[j * 3 + 1];
  float dz = pos[i * 3 + 2] - pos[j * 3 + 2];
  float dist = sqrtf(dx * dx + dy * dy + dz * dz);
  float in4[4] = {dx, dy, dz, dist};
  float out = bd2[0];
#pragma unroll
  for (int jj = 0; jj < 32; ++jj) {
    float s = bd1[jj];
#pragma unroll
    for (int ii = 0; ii < 4; ++ii) s += in4[ii] * Wd1[ii * 32 + jj];
    out += fmaxf(s, 0.f) * Wd2[jj];
  }
  dmask[e] = 1.f / (1.f + expf(-out));
}

// monotone float max via signed-max / unsigned-min (order-independent => deterministic)
__device__ __forceinline__ void atomicMaxF(float* a, float v) {
  if (v >= 0.f) atomicMax((int*)a, __float_as_int(v));
  else          atomicMin((unsigned int*)a, __float_as_uint(v));
}

// in-place dual softmax + KL + message scatter-max. one thread per (e,h).
__global__ __launch_bounds__(256) void softmax_kl_msg_k(
    float* p3, float* ptx, float* klout, const float* V, const float* dmask,
    const int* row, float* agg, int E) {
  long long gid = blockIdx.x * 256LL + threadIdx.x;
  if (gid >= (long long)E * HDIM) return;     // E*8 is a multiple of 256: no divergence
  int e = (int)(gid >> 3), h = (int)(gid & 7);
  float s3 = dmask[e] * INV_TEMP, st = INV_TEMP;
  float l3[32], lt[32];
  float m3 = -1e30f, mt = -1e30f;
#pragma unroll
  for (int c = 0; c < 32; ++c) {
    l3[c] = p3[((long long)e * 32 + c) * HDIM + h] * s3;
    lt[c] = ptx[((long long)e * 32 + c) * HDIM + h] * st;
    m3 = fmaxf(m3, l3[c]); mt = fmaxf(mt, lt[c]);
  }
  float z3 = 0.f, zt = 0.f;
#pragma unroll
  for (int c = 0; c < 32; ++c) { z3 += expf(l3[c] - m3); zt += expf(lt[c] - mt); }
  float ln3 = m3 + logf(z3), lnt = mt + logf(zt);
  float kl = 0.f;
  int rw = row[e];
#pragma unroll
  for (int c = 0; c < 32; ++c) {
    float lp3 = l3[c] - ln3, lpt = lt[c] - lnt;
    float p3v = expf(lp3), ptv = expf(lpt);
    p3[((long long)e * 32 + c) * HDIM + h] = p3v;
    ptx[((long long)e * 32 + c) * HDIM + h] = ptv;
    kl += p3v * (lp3 - lpt);
    float msg = 0.5f * (p3v + ptv) * V[(long long)e * 256 + c * HDIM + h];
    atomicMaxF(&agg[(long long)rw * 256 + c * HDIM + h], msg);
  }
  kl += __shfl_xor(kl, 1, 32);
  kl += __shfl_xor(kl, 2, 32);
  kl += __shfl_xor(kl, 4, 32);
  if (h == 0) klout[e] = kl * 0.125f;
}

__global__ void agg_fin_k(float* agg, long long n) {
  long long i = blockIdx.x * 256LL + threadIdx.x; if (i >= n) return;
  float v = agg[i];
  unsigned u = __float_as_uint(v) & 0x7f800000u;
  agg[i] = (u == 0x7f800000u) ? 0.f : v;      // -inf (no edges) -> 0
}

__global__ void edge_scatter_k(const float* ue, const int* row, const int* col,
                               float* osum, float* isum, float* ocnt, float* icnt, int E) {
  long long gid = blockIdx.x * 256LL + threadIdx.x;
  if (gid >= (long long)E * 256) return;
  int e = (int)(gid >> 8), j = (int)(gid & 255);
  float v = ue[(long long)e * 256 + j];
  atomicAdd(&osum[(long long)row[e] * 256 + j], v);
  atomicAdd(&isum[(long long)col[e] * 256 + j], v);
  if (j == 0) { atomicAdd(&ocnt[row[e]], 1.f); atomicAdd(&icnt[col[e]], 1.f); }
}

__global__ void twin_k(const float* osum, const float* isum, const float* ocnt, const float* icnt,
                       float* twin, int Nn) {
  long long gid = blockIdx.x * 256LL + threadIdx.x;
  if (gid >= (long long)Nn * 256) return;
  int n = (int)(gid >> 8), j = (int)(gid & 255);
  twin[(long long)n * 512 + j]       = osum[gid] / fmaxf(ocnt[n], 1.f);
  twin[(long long)n * 512 + 256 + j] = isum[gid] / fmaxf(icnt[n], 1.f);
}

__global__ void final_k(const float* un, const float* ea, float* fn, long long n) {
  long long i = blockIdx.x * 256LL + threadIdx.x; if (i >= n) return;
  fn[i] = fmaxf(un[i], 0.f) * ea[i];
}

// ------------------------------- host ---------------------------------------

extern "C" void kernel_launch(void* const* d_in, const int* in_sizes, int n_in,
                              void* d_out, int out_size, void* d_ws, size_t ws_size,
                              hipStream_t stream) {
  (void)n_in; (void)out_size; (void)ws_size;
  const int Nn = in_sizes[0] / 256;     // 20000
  const int E  = in_sizes[2] / 2;      // 160000

  const float* x    = (const float*)d_in[0];
  const float* ef   = (const float*)d_in[1];
  const int*   eidx = (const int*)d_in[2];   // int32 per harness convention
  const float* pos  = (const float*)d_in[3];
  const float* text = (const float*)d_in[4];
  const float* Wq  = (const float*)d_in[5];  const float* bq  = (const float*)d_in[6];
  const float* Wk  = (const float*)d_in[7];  const float* bk  = (const float*)d_in[8];
  const float* Wv  = (const float*)d_in[9];  const float* bv  = (const float*)d_in[10];
  const float* Wt  = (const float*)d_in[11]; const float* bt  = (const float*)d_in[12];
  const float* Wd1 = (const float*)d_in[13]; const float* bd1 = (const float*)d_in[14];
  const float* Wd2 = (const float*)d_in[15]; const float* bd2 = (const float*)d_in[16];
  const float* We1 = (const float*)d_in[17]; const float* be1 = (const float*)d_in[18];
  const float* We2 = (const float*)d_in[19]; const float* be2 = (const float*)d_in[20];
  const float* Wea = (const float*)d_in[21]; const float* bea = (const float*)d_in[22];
  const float* Wn1 = (const float*)d_in[23]; const float* bn1 = (const float*)d_in[24];
  const float* Wn2 = (const float*)d_in[25]; const float* bn2 = (const float*)d_in[26];
  const float* Wa1 = (const float*)d_in[27]; const float* ba1 = (const float*)d_in[28];
  const float* Wa2 = (const float*)d_in[29]; const float* ba2 = (const float*)d_in[30];
  const float* Wx1 = (const float*)d_in[31]; const float* bx1 = (const float*)d_in[32];
  const float* Wx2 = (const float*)d_in[33]; const float* bx2 = (const float*)d_in[34];

  const int* row = eidx;
  const int* col = eidx + E;

  // d_out layout (return order): final_node, updated_edge, p3, ptx, kl
  float* out_fn = (float*)d_out;
  float* out_ue = out_fn + (long long)Nn * 256;
  float* out_p3 = out_ue + (long long)E * 256;
  float* out_pt = out_p3 + (long long)E * 256;
  float* out_kl = out_pt + (long long)E * 256;

  // workspace carve-out
  char* w = (char*)d_ws; size_t off = 0;
  auto alloc = [&](size_t bytes) -> void* {
    void* p = w + off; off += (bytes + 255) & ~(size_t)255; return p;
  };
  unsigned* tk   = (unsigned*)alloc(HSZ * 4);
  unsigned* tv   = (unsigned*)alloc(HSZ * 4);
  int*      rev  = (int*)alloc((size_t)E * 4);
  float*    dmk  = (float*)alloc((size_t)E * 4);
  float*    ptb  = (float*)alloc((size_t)E * 256 * 4);
  float*    Qb   = (float*)alloc((size_t)E * 256 * 4);
  float*    Kb   = (float*)alloc((size_t)E * 256 * 4);
  float*    Vb   = (float*)alloc((size_t)E * 256 * 4);
  float*    h1b  = (float*)alloc((size_t)E * 768 * 4);   // reused for h3 / ht
  float*    agg  = (float*)alloc((size_t)Nn * 256 * 4);
  float*    hnb  = (float*)alloc((size_t)Nn * 512 * 4);
  float*    unb  = (float*)alloc((size_t)Nn * 256 * 4);
  float*    twin = (float*)alloc((size_t)Nn * 512 * 4);
  float*    eab  = (float*)alloc((size_t)Nn * 256 * 4);
  float*    osum = (float*)alloc((size_t)Nn * 256 * 4);
  float*    isum = (float*)alloc((size_t)Nn * 256 * 4);
  float*    ocnt = (float*)alloc((size_t)Nn * 4);
  float*    icnt = (float*)alloc((size_t)Nn * 4);

  auto gemm = [&](GemmDesc d) {
    dim3 g((unsigned)((d.M + BM - 1) / BM), (unsigned)((d.N + BN - 1) / BN));
    gemm_bf16_wmma<<<g, dim3(256), 0, stream>>>(d);
  };
  auto blocks = [](long long n) { return dim3((unsigned)((n + 255) / 256)); };

  // 1) reverse-edge hash
  fill_u32_k<<<blocks(HSZ), 256, 0, stream>>>(tk, HSZ, 0xFFFFFFFFu);
  fill_u32_k<<<blocks(HSZ), 256, 0, stream>>>(tv, HSZ, 0xFFFFFFFFu);
  hash_insert_k<<<blocks(E), 256, 0, stream>>>(row, col, tk, tv, Nn, E);
  hash_lookup_k<<<blocks(E), 256, 0, stream>>>(row, col, tk, tv, rev, Nn, E);

  // 2) distance mask
  dmask_k<<<blocks(E), 256, 0, stream>>>(pos, row, col, Wd1, bd1, Wd2, bd2, dmk, E);

  // 3) dense projections (WMMA)
  { GemmDesc d{}; d.A0 = text; d.B = Wt; d.bias = bt; d.C = ptb; d.M = E; d.N = 256; d.K = 512; d.segw = 512; gemm(d); }
  { GemmDesc d{}; d.A0 = x; d.I0 = row; d.B = Wq; d.bias = bq; d.C = Qb; d.M = E; d.N = 256; d.K = 256; d.segw = 256; gemm(d); }
  { GemmDesc d{}; d.A0 = ef;            d.B = Wk; d.bias = bk; d.C = Kb; d.M = E; d.N = 256; d.K = 256; d.segw = 256; gemm(d); }
  { GemmDesc d{}; d.A0 = x; d.I0 = col; d.B = Wv; d.bias = bv; d.C = Vb; d.M = E; d.N = 256; d.K = 256; d.segw = 256; gemm(d); }

  // 4) edge MLP: concat[x_i, ef, rev_ef, x_j] -> relu -> updated_edge
  { GemmDesc d{}; d.A0 = x; d.I0 = row; d.A1 = ef; d.A2 = ef; d.I2 = rev; d.A3 = x; d.I3 = col;
    d.B = We1; d.bias = be1; d.C = h1b; d.M = E; d.N = 768; d.K = 1024; d.segw = 256; d.act = 1; gemm(d); }
  { GemmDesc d{}; d.A0 = h1b; d.B = We2; d.bias = be2; d.C = out_ue; d.M = E; d.N = 256; d.K = 768; d.segw = 768; gemm(d); }

  // 5) 3D-attention chan-MLP (head-strided rows, W stored [O][C] -> btrans)
  { GemmDesc d{}; d.A0 = Qb; d.A1 = Kb; d.B = Wa1; d.bias = ba1; d.C = h1b;
    d.M = E * HDIM; d.N = 64; d.K = 64; d.segw = 32; d.hstride = 1; d.outh = 1; d.btrans = 1; d.act = 1; gemm(d); }
  { GemmDesc d{}; d.A0 = h1b; d.B = Wa2; d.bias = ba2; d.C = out_p3;
    d.M = E * HDIM; d.N = 32; d.K = 64; d.segw = 64; d.hstride = 1; d.outh = 1; d.btrans = 1; gemm(d); }

  // 6) text-guided chan-MLP
  { GemmDesc d{}; d.A0 = Qb; d.A1 = Kb; d.A2 = ptb; d.B = Wx1; d.bias = bx1; d.C = h1b;
    d.M = E * HDIM; d.N = 96; d.K = 96; d.segw = 32; d.hstride = 1; d.outh = 1; d.btrans = 1; d.act = 1; gemm(d); }
  { GemmDesc d{}; d.A0 = h1b; d.B = Wx2; d.bias = bx2; d.C = out_pt;
    d.M = E * HDIM; d.N = 32; d.K = 96; d.segw = 96; d.hstride = 1; d.outh = 1; d.btrans = 1; gemm(d); }

  // 7) softmaxes + KL + scatter-max aggregation
  fill_f32_k<<<blocks((long long)Nn * 256), 256, 0, stream>>>(agg, (long long)Nn * 256, -__builtin_inff());
  softmax_kl_msg_k<<<blocks((long long)E * HDIM), 256, 0, stream>>>(out_p3, out_pt, out_kl, Vb, dmk, row, agg, E);
  agg_fin_k<<<blocks((long long)Nn * 256), 256, 0, stream>>>(agg, (long long)Nn * 256);

  // 8) node MLP on concat[x, agg]
  { GemmDesc d{}; d.A0 = x; d.A1 = agg; d.B = Wn1; d.bias = bn1; d.C = hnb;
    d.M = Nn; d.N = 512; d.K = 512; d.segw = 256; d.act = 1; gemm(d); }
  { GemmDesc d{}; d.A0 = hnb; d.B = Wn2; d.bias = bn2; d.C = unb;
    d.M = Nn; d.N = 256; d.K = 512; d.segw = 512; gemm(d); }

  // 9) twin edge attention: per-node mean of outgoing/incoming updated edges
  fill_f32_k<<<blocks((long long)Nn * 256), 256, 0, stream>>>(osum, (long long)Nn * 256, 0.f);
  fill_f32_k<<<blocks((long long)Nn * 256), 256, 0, stream>>>(isum, (long long)Nn * 256, 0.f);
  fill_f32_k<<<blocks(Nn), 256, 0, stream>>>(ocnt, Nn, 0.f);
  fill_f32_k<<<blocks(Nn), 256, 0, stream>>>(icnt, Nn, 0.f);
  edge_scatter_k<<<blocks((long long)E * 256), 256, 0, stream>>>(out_ue, row, col, osum, isum, ocnt, icnt, E);
  twin_k<<<blocks((long long)Nn * 256), 256, 0, stream>>>(osum, isum, ocnt, icnt, twin, Nn);
  { GemmDesc d{}; d.A0 = twin; d.B = Wea; d.bias = bea; d.C = eab;
    d.M = Nn; d.N = 256; d.K = 512; d.segw = 512; d.act = 2; gemm(d); }

  // 10) final node output
  final_k<<<blocks((long long)Nn * 256), 256, 0, stream>>>(unb, eab, out_fn, (long long)Nn * 256);
}
